// QRNN_1bit_44633300140405
// MI455X (gfx1250) — 8-bit matrix operand order repaired and checked by reading, compiled, not run on hardware
//
#include <hip/hip_runtime.h>
#include <hip/hip_bf16.h>

#define VOCAB 32000
#define E_DIM 256
#define H_DIM 512
#define T_DIM 512
#define B_DIM 64

typedef __attribute__((ext_vector_type(8))) int v8i;
typedef int v4i_vec __attribute__((vector_size(16)));

#ifndef __has_builtin
#define __has_builtin(x) 0
#endif
#if __has_builtin(__builtin_amdgcn_global_load_async_to_lds_b128) && \
    __has_builtin(__builtin_amdgcn_s_wait_asynccnt)
#define USE_ASYNC_LDS 1
#else
#define USE_ASYNC_LDS 0
#endif

__device__ __forceinline__ void copy16_g2lds(void* lds_dst, const void* gsrc) {
#if USE_ASYNC_LDS
  __builtin_amdgcn_global_load_async_to_lds_b128(
      (__attribute__((address_space(1))) v4i_vec*)(unsigned long long)gsrc,
      (__attribute__((address_space(3))) v4i_vec*)(unsigned)(unsigned long long)lds_dst,
      0, 0);
#else
  *(int4*)lds_dst = *(const int4*)gsrc;
#endif
}

__device__ __forceinline__ void wait_async_copies() {
#if USE_ASYNC_LDS
  __builtin_amdgcn_s_wait_asynccnt(0);
#endif
}

__device__ __forceinline__ v8i load_a_frag(const signed char* base, int lda, int lane) {
  const int m  = lane & 15;
  const int kg = (lane >> 4) * 8;
  const signed char* row = base + m * lda;
  v8i a;
#pragma unroll
  for (int v = 0; v < 8; ++v) {
    int k = kg + (v & 1) * 4 + ((v >> 1) & 1) * 16 + (v >> 2) * 32;
    a[v] = *(const int*)(row + k);
  }
  return a;
}

__device__ __forceinline__ v8i load_b_frag(const signed char* base, int ldb, int lane) {
  const int n  = lane & 15;
  const int kg = (lane >> 4) * 8;
  const signed char* row = base + n * ldb;
  v8i b;
#pragma unroll
  for (int v = 0; v < 8; ++v) {
    int k = kg + (v & 1) * 4 + ((v >> 1) & 1) * 16 + (v >> 2) * 32;
    b[v] = *(const int*)(row + k);
  }
  return b;
}

__global__ void k_zero_u32(unsigned* p) { *p = 0u; }

__global__ void k_amax(const float* __restrict__ x, unsigned* amax_bits, int n) {
  unsigned local = 0u;
  for (int i = blockIdx.x * blockDim.x + threadIdx.x; i < n; i += gridDim.x * blockDim.x) {
    local = max(local, __float_as_uint(fabsf(x[i])));
  }
  atomicMax(amax_bits, local);
}

__global__ void k_sign_pack(const float* __restrict__ src, signed char* __restrict__ dst, int n) {
  int i = blockIdx.x * blockDim.x + threadIdx.x;
  if (i < n) dst[i] = (src[i] >= 0.0f) ? (signed char)1 : (signed char)-1;
}

__global__ void k_add2(const float* __restrict__ a, const float* __restrict__ b,
                       float* __restrict__ o, int n) {
  int i = blockIdx.x * blockDim.x + threadIdx.x;
  if (i < n) o[i] = a[i] + b[i];
}

__global__ void k_embed_gather(const int* __restrict__ text, const float* __restrict__ emb,
                               const unsigned* __restrict__ amax_bits,
                               signed char* __restrict__ dst) {
  long idx = (long)blockIdx.x * blockDim.x + threadIdx.x;
  const long total = (long)T_DIM * B_DIM * E_DIM;
  if (idx >= total) return;
  int e = (int)(idx % E_DIM);
  long tb = idx / E_DIM;
  int b = (int)(tb % B_DIM);
  int t = (int)(tb / B_DIM);
  float amax = __uint_as_float(*amax_bits) + 1e-12f;
  float s = exp2f(ceilf(log2f(amax / 127.0f)));
  int tok = text[b * T_DIM + t];
  float v = emb[(long)tok * E_DIM + e];
  dst[idx] = (v >= -0.5f * s) ? (signed char)1 : (signed char)-1;
}

__global__ void __launch_bounds__(256) k_gemm_bin(const signed char* __restrict__ A,
                                                  const signed char* __restrict__ W,
                                                  const float* __restrict__ bias,
                                                  float* __restrict__ C,
                                                  int M, int N, int K) {
  const int wave = threadIdx.x >> 5;
  const int lane = threadIdx.x & 31;
  const int tile = blockIdx.x * (blockDim.x >> 5) + wave;
  const int ntn  = N >> 4;
  if (tile >= (M >> 4) * ntn) return;
  const int mt = tile / ntn;
  const int nt = tile % ntn;

  const signed char* Abase = A + (long)mt * 16 * K;
  const signed char* Wbase = W + (long)nt * 16 * K;

  v8i acc = {};
  v8i a0 = load_a_frag(Abase, K, lane);
  v8i b0 = load_b_frag(Wbase, K, lane);
  for (int kk = 64; kk < K; kk += 64) {
    __builtin_prefetch(Abase + (long)(lane & 15) * K + kk + 64, 0, 3);
    v8i a1 = load_a_frag(Abase + kk, K, lane);
    v8i b1 = load_b_frag(Wbase + kk, K, lane);
    acc = __builtin_amdgcn_wmma_i32_16x16x64_iu8(true, a0, true, b0, acc, false, false);
    a0 = a1;
    b0 = b1;
  }
  acc = __builtin_amdgcn_wmma_i32_16x16x64_iu8(true, a0, true, b0, acc, false, false);

  const int n  = lane & 15;
  const int gn = nt * 16 + n;
  const float bv = bias[gn];
#pragma unroll
  for (int v = 0; v < 8; ++v) {
    int gm = mt * 16 + v + ((lane >> 4) << 3);
    C[(long)gm * N + gn] = (float)acc[v] + bv;
  }
}

__global__ void __launch_bounds__(1024) k_recur(const signed char* __restrict__ Whh_s,
                                                const float* __restrict__ xW,
                                                signed char* __restrict__ ys,
                                                int storeAll) {
  extern __shared__ char smem[];
  signed char* whh = (signed char*)smem;
  signed char* hA  = (signed char*)(smem + H_DIM * H_DIM);
  signed char* hB  = hA + B_DIM * H_DIM;

  const int tid = threadIdx.x;
  {
    const int4* src = (const int4*)Whh_s;
    int4* dst = (int4*)whh;
    for (int i = tid; i < (H_DIM * H_DIM) / 16; i += 1024)
      copy16_g2lds(dst + i, src + i);
    int4 z; z.x = 0; z.y = 0; z.z = 0; z.w = 0;
    int4* hz = (int4*)hA;
    for (int i = tid; i < (B_DIM * H_DIM) / 16; i += 1024) hz[i] = z;
    wait_async_copies();
  }
  __syncthreads();

  const int wave = tid >> 5;
  const int lane = tid & 31;
  const int tile0 = wave * 4;
  const int mt  = tile0 >> 5;
  const int nt0 = tile0 & 31;

  for (int t = 0; t < T_DIM; ++t) {
    signed char* hprev = (t & 1) ? hB : hA;
    signed char* hnext = (t & 1) ? hA : hB;

    if (t + 1 < T_DIM) {
      const float* nxt = xW + (long)(t + 1) * B_DIM * H_DIM +
                         (long)(mt * 16 + (lane & 15)) * H_DIM + nt0 * 16 +
                         (lane >> 4) * 32;
      __builtin_prefetch(nxt, 0, 3);
    }

    v8i acc[4] = {};
    const signed char* Ab = hprev + (long)mt * 16 * H_DIM;
    for (int kk = 0; kk < H_DIM; kk += 64) {
      v8i a = load_a_frag(Ab + kk, H_DIM, lane);
#pragma unroll
      for (int i = 0; i < 4; ++i) {
        v8i b = load_b_frag(whh + (long)(nt0 + i) * 16 * H_DIM + kk, H_DIM, lane);
        acc[i] = __builtin_amdgcn_wmma_i32_16x16x64_iu8(true, a, true, b, acc[i],
                                                        false, false);
      }
    }

    const int n  = lane & 15;
    const int mh = (lane >> 4) << 3;
    const long rowbase = (long)t * B_DIM * H_DIM;
#pragma unroll
    for (int i = 0; i < 4; ++i) {
      int gn = (nt0 + i) * 16 + n;
#pragma unroll
      for (int v = 0; v < 8; ++v) {
        int gm = mt * 16 + v + mh;
        float val = (float)acc[i][v] + xW[rowbase + (long)gm * H_DIM + gn];
        signed char sv = (val >= 0.0f) ? (signed char)1 : (signed char)-1;
        hnext[gm * H_DIM + gn] = sv;
        if (storeAll || t == T_DIM - 1)
          ys[rowbase + (long)gm * H_DIM + gn] = sv;
      }
    }
    __syncthreads();
  }
}

__global__ void k_dense(const signed char* __restrict__ ys0,
                        const signed char* __restrict__ ys1,
                        const signed char* __restrict__ wfc,
                        float* __restrict__ out) {
  int id = threadIdx.x;
  if (id >= 2 * B_DIM) return;
  const signed char* ys = (id < B_DIM) ? ys0 : ys1;
  int b = id & (B_DIM - 1);
  long row = ((long)(T_DIM - 1) * B_DIM + b) * H_DIM;
  int sum = 0;
  for (int h = 0; h < H_DIM; ++h) sum += (int)ys[row + h] * (int)wfc[h];
  out[id] = (float)sum;
}

extern "C" void kernel_launch(void* const* d_in, const int* in_sizes, int n_in,
                              void* d_out, int out_size, void* d_ws, size_t ws_size,
                              hipStream_t stream) {
  (void)in_sizes; (void)n_in; (void)out_size; (void)ws_size;
  const int*   text = (const int*)  d_in[0];
  const float* emb  = (const float*)d_in[2];
  const float* Wih0 = (const float*)d_in[3];
  const float* Whh0 = (const float*)d_in[4];
  const float* bih0 = (const float*)d_in[5];
  const float* bhh0 = (const float*)d_in[6];
  const float* Wih1 = (const float*)d_in[7];
  const float* Whh1 = (const float*)d_in[8];
  const float* bih1 = (const float*)d_in[9];
  const float* bhh1 = (const float*)d_in[10];
  const float* Wfc  = (const float*)d_in[11];
  float* out = (float*)d_out;

  char* ws = (char*)d_ws;
  size_t off = 0;
  auto take = [&](size_t sz) -> char* {
    char* p = ws + off;
    off += (sz + 255) & ~(size_t)255;
    return p;
  };
  unsigned*    amax  = (unsigned*)   take(4);
  signed char* Wih0s = (signed char*)take((size_t)H_DIM * E_DIM);
  signed char* Whh0s = (signed char*)take((size_t)H_DIM * H_DIM);
  signed char* Wih1s = (signed char*)take((size_t)H_DIM * H_DIM);
  signed char* Whh1s = (signed char*)take((size_t)H_DIM * H_DIM);
  signed char* Wfcs  = (signed char*)take((size_t)H_DIM);
  float*       b0    = (float*)      take((size_t)H_DIM * 4);
  float*       b1    = (float*)      take((size_t)H_DIM * 4);
  signed char* xsign = (signed char*)take((size_t)T_DIM * B_DIM * E_DIM);
  signed char* ys0   = (signed char*)take((size_t)T_DIM * B_DIM * H_DIM);
  signed char* ys1   = (signed char*)take((size_t)T_DIM * B_DIM * H_DIM);
  float*       xW0   = (float*)      take((size_t)T_DIM * B_DIM * H_DIM * 4);
  float*       xW1   = (float*)      take((size_t)T_DIM * B_DIM * H_DIM * 4);

  k_zero_u32<<<1, 1, 0, stream>>>(amax);
  k_amax<<<512, 256, 0, stream>>>(emb, amax, VOCAB * E_DIM);

  k_sign_pack<<<(H_DIM * E_DIM + 255) / 256, 256, 0, stream>>>(Wih0, Wih0s, H_DIM * E_DIM);
  k_sign_pack<<<(H_DIM * H_DIM + 255) / 256, 256, 0, stream>>>(Whh0, Whh0s, H_DIM * H_DIM);
  k_sign_pack<<<(H_DIM * H_DIM + 255) / 256, 256, 0, stream>>>(Wih1, Wih1s, H_DIM * H_DIM);
  k_sign_pack<<<(H_DIM * H_DIM + 255) / 256, 256, 0, stream>>>(Whh1, Whh1s, H_DIM * H_DIM);
  k_sign_pack<<<(H_DIM + 255) / 256, 256, 0, stream>>>(Wfc, Wfcs, H_DIM);
  k_add2<<<(H_DIM + 255) / 256, 256, 0, stream>>>(bih0, bhh0, b0, H_DIM);
  k_add2<<<(H_DIM + 255) / 256, 256, 0, stream>>>(bih1, bhh1, b1, H_DIM);

  {
    long total = (long)T_DIM * B_DIM * E_DIM;
    k_embed_gather<<<(unsigned)((total + 255) / 256), 256, 0, stream>>>(text, emb, amax, xsign);
  }

  {
    int tiles = (T_DIM * B_DIM / 16) * (H_DIM / 16);
    k_gemm_bin<<<tiles / 8, 256, 0, stream>>>(xsign, Wih0s, b0, xW0,
                                              T_DIM * B_DIM, H_DIM, E_DIM);
  }

  k_recur<<<1, 1024, 320 * 1024, stream>>>(Whh0s, xW0, ys0, 1);

  {
    int tiles = (T_DIM * B_DIM / 16) * (H_DIM / 16);
    k_gemm_bin<<<tiles / 8, 256, 0, stream>>>(ys0, Wih1s, b1, xW1,
                                              T_DIM * B_DIM, H_DIM, H_DIM);
  }

  k_recur<<<1, 1024, 320 * 1024, stream>>>(Whh1s, xW1, ys1, 0);

  k_dense<<<1, 128, 0, stream>>>(ys0, ys1, Wfcs, out);
}
